// RelPartialLearnableMultiHeadAttn_14139032338662
// MI455X (gfx1250) — matrix operand order repaired and checked by reading, not compiled, not run on hardware
//
#include <hip/hip_runtime.h>
#include <hip/hip_bf16.h>
#include <stdint.h>

#define T_SEQ  1024
#define BATCH  8
#define DMODEL 1024
#define NH     16
#define DH     64
#define TB     (T_SEQ * BATCH)

typedef __attribute__((ext_vector_type(16))) __bf16 v16bf;
typedef __attribute__((ext_vector_type(8)))  float  v8f;

#if __has_builtin(__builtin_amdgcn_global_load_async_to_lds_b128)
#define HAVE_ASYNC_LDS 1
#else
#define HAVE_ASYNC_LDS 0
#endif
#if __has_builtin(__builtin_amdgcn_tensor_load_to_lds)
#define HAVE_TDM 1
#else
#define HAVE_TDM 0
#endif

union Frag16 {
  v16bf v;
  unsigned u[8];
  unsigned short h[16];
};

__device__ __forceinline__ unsigned short f2bf(float f) {
  unsigned u = __float_as_uint(f);
  u += 0x7FFFu + ((u >> 16) & 1u);       // round-to-nearest-even
  return (unsigned short)(u >> 16);
}

// A-fragment (16x32, 16-bit) K index for half-pair p (0..7), lane-half hi.
__device__ __forceinline__ int akp(int p, int hi) {
  return ((p & 3) << 1) + ((p >> 2) << 4) + (hi ? 8 : 0);
}
// B-fragment (32x16, 16-bit) K index: same K order as the A-fragment.
__device__ __forceinline__ int bkp(int p, int hi) {
  return ((p & 3) << 1) + ((p >> 2) << 4) + (hi ? 8 : 0);
}

__device__ __forceinline__ v8f wmma_bf(const Frag16& a, const Frag16& b, v8f c) {
  return __builtin_amdgcn_wmma_f32_16x16x32_bf16(false, a.v, false, b.v,
                                                 (short)0, c, false, false);
}

#if HAVE_ASYNC_LDS
typedef int __attribute__((ext_vector_type(4))) i4vec;
// per-lane 16B async copy global -> LDS (ASYNCcnt)
__device__ __forceinline__ void async_cp16B(const unsigned short* g,
                                            unsigned short* l) {
  __builtin_amdgcn_global_load_async_to_lds_b128(
      (__attribute__((address_space(1))) i4vec*)g,
      (__attribute__((address_space(3))) i4vec*)l, 0, 0);
}
#endif

#if HAVE_TDM
typedef unsigned __attribute__((ext_vector_type(4))) v4u;
typedef int      __attribute__((ext_vector_type(8))) v8i;
typedef int      __attribute__((ext_vector_type(4))) v4i;
// 2D TDM tile load: tile (tileK x tileN rows) of bf16 from a row-major
// [ntot x ktot] tensor (row stride = strideK elements) into LDS at ldsAddr.
__device__ __forceinline__ void tdm_load_2d(unsigned ldsAddr, const void* gptr,
                                            unsigned tileK, unsigned tileN,
                                            unsigned ktot, unsigned ntot,
                                            unsigned strideK) {
  unsigned long long ga = (unsigned long long)gptr;
  v4u g0;
  g0[0] = 1u;                                            // count=1, user mode
  g0[1] = ldsAddr;                                       // lds_addr
  g0[2] = (unsigned)ga;                                  // global_addr[31:0]
  g0[3] = (unsigned)((ga >> 32) & 0x1FFFFFFull) | (2u << 30);  // addr hi, type=2
  v8i g1;
  g1[0] = (int)(1u << 16);                               // data_size=1 (2B), mask=0
  g1[1] = (int)((ktot & 0xFFFFu) << 16);                 // tensor_dim0 lo
  g1[2] = (int)((ktot >> 16) | ((ntot & 0xFFFFu) << 16));// tensor_dim0 hi, dim1 lo
  g1[3] = (int)((ntot >> 16) | (tileK << 16));           // dim1 hi, tile_dim0
  g1[4] = (int)(tileN & 0xFFFFu);                        // tile_dim1, tile_dim2=0
  g1[5] = (int)strideK;                                  // tensor_dim0_stride lo
  g1[6] = 0;                                             // stride hi, dim1_stride lo
  g1[7] = 0;
  v4i gz = {0, 0, 0, 0};
#if __clang_major__ >= 23
  v8i gz8 = {0, 0, 0, 0, 0, 0, 0, 0};
  __builtin_amdgcn_tensor_load_to_lds(g0, g1, gz, gz, gz8, 0);
#else
  __builtin_amdgcn_tensor_load_to_lds(g0, g1, gz, gz, 0);
#endif
}
#endif

// ---------------------------------------------------------------- pack kernels
__global__ void pack_x_kernel(const float* __restrict__ in,
                              const float* __restrict__ pe,
                              unsigned short* __restrict__ xb, int n) {
  int i = blockIdx.x * 256 + threadIdx.x;
  if (i < n) xb[i] = f2bf(in[i] + pe[i]);
}

// Wt[n][k] = bf16(W[k][n])  (W is K x N row-major)
__global__ void pack_wt_kernel(const float* __restrict__ w,
                               unsigned short* __restrict__ wt, int K, int N) {
  int o = blockIdx.x * 256 + threadIdx.x;
  if (o < K * N) {
    int ncol = o / K, k = o - ncol * K;
    wt[o] = f2bf(w[(size_t)k * N + ncol]);
  }
}

__global__ void zero_kernel(float* __restrict__ p, int n) {
  int i = blockIdx.x * 256 + threadIdx.x;
  if (i < n) p[i] = 0.0f;
}

// ------------------------------------------------------------- GEMM tile core
// C(128x64) tile of A[M x K] * Bt[N x K]^T ; 256 threads = 8 waves (4x2 of 32x32)
// Double-buffered LDS pipeline: async/TDM staged tile k+1 overlaps WMMAs on k.
template <int KTOT>
__device__ __forceinline__ void gemm_core(const unsigned short* __restrict__ Ag,
                                          const unsigned short* __restrict__ Btg,
                                          int rowBase, int colBase, int ntot,
                                          v8f acc[2][2]) {
  __shared__ unsigned short As[2][128][32];
  __shared__ unsigned short Bs[2][64][32];
  const int tid  = threadIdx.x;
  const int wave = tid >> 5, lane = tid & 31, ln = lane & 15;
  const int hi   = lane >= 16;
  const int wm   = wave >> 1, wn = wave & 1;
  const int alr  = tid >> 1, aseg = (tid & 1) * 16;
  const int blr  = tid >> 2, bseg = (tid & 3) * 8;
  (void)ntot; (void)blr; (void)bseg;

  auto compute = [&](int buf) {
    Frag16 af[2], bf2[2];
#pragma unroll
    for (int mi = 0; mi < 2; ++mi) {
      int r = wm * 32 + mi * 16 + ln;
#pragma unroll
      for (int p = 0; p < 8; ++p)
        af[mi].u[p] = *(const unsigned*)&As[buf][r][akp(p, hi)];
    }
#pragma unroll
    for (int ni = 0; ni < 2; ++ni) {
      int c = wn * 32 + ni * 16 + ln;
#pragma unroll
      for (int p = 0; p < 8; ++p)
        bf2[ni].u[p] = *(const unsigned*)&Bs[buf][c][bkp(p, hi)];
    }
#pragma unroll
    for (int mi = 0; mi < 2; ++mi)
#pragma unroll
      for (int ni = 0; ni < 2; ++ni)
        acc[mi][ni] = wmma_bf(af[mi], bf2[ni], acc[mi][ni]);
  };

#if HAVE_ASYNC_LDS
  auto stage = [&](int k0, int buf) {
    const unsigned short* aSrc = &Ag[(size_t)(rowBase + alr) * KTOT + k0 + aseg];
    async_cp16B(aSrc,     &As[buf][alr][aseg]);
    async_cp16B(aSrc + 8, &As[buf][alr][aseg + 8]);
#if HAVE_TDM
    if (wave == 0) {
      unsigned laddr = (unsigned)(unsigned long long)
          (__attribute__((address_space(3))) void*)&Bs[buf][0][0];
      tdm_load_2d(laddr, &Btg[(size_t)colBase * KTOT + k0],
                  32u, 64u, (unsigned)KTOT, (unsigned)ntot, (unsigned)KTOT);
    }
#else
    async_cp16B(&Btg[(size_t)(colBase + blr) * KTOT + k0 + bseg],
                &Bs[buf][blr][bseg]);
#endif
  };

  constexpr int nk = KTOT / 32;
  stage(0, 0);
  for (int k = 0; k < nk; ++k) {
    const int cur = k & 1;
    if (k + 1 < nk) stage((k + 1) * 32, 1 - cur);       // overlap next tile
#if HAVE_TDM
    if (k + 1 < nk) asm volatile("s_wait_asynccnt 0x2" ::: "memory");
    else            asm volatile("s_wait_asynccnt 0x0" ::: "memory");
    if (wave == 0) {
      if (k + 1 < nk) asm volatile("s_wait_tensorcnt 0x1" ::: "memory");
      else            asm volatile("s_wait_tensorcnt 0x0" ::: "memory");
    }
#else
    if (k + 1 < nk) asm volatile("s_wait_asynccnt 0x3" ::: "memory");
    else            asm volatile("s_wait_asynccnt 0x0" ::: "memory");
#endif
    __syncthreads();
    compute(cur);
    __syncthreads();
  }
#else
  // fallback: synchronous staging, single buffer
  for (int k0 = 0; k0 < KTOT; k0 += 32) {
    const unsigned short* aSrc = &Ag[(size_t)(rowBase + alr) * KTOT + k0 + aseg];
    const unsigned short* bSrc = &Btg[(size_t)(colBase + blr) * KTOT + k0 + bseg];
    *(uint4*)&As[0][alr][aseg]     = *(const uint4*)aSrc;
    *(uint4*)&As[0][alr][aseg + 8] = *(const uint4*)(aSrc + 8);
    *(uint4*)&Bs[0][blr][bseg]     = *(const uint4*)bSrc;
    if (k0 + 32 < KTOT) {
      __builtin_prefetch(aSrc + 32, 0, 1);
      __builtin_prefetch(bSrc + 32, 0, 1);
    }
    __syncthreads();
    compute(0);
    __syncthreads();
  }
#endif
}

// --------------------------------------------------------------- QKV GEMM
__global__ __launch_bounds__(256)
void gemm_qkv_kernel(const unsigned short* __restrict__ Xb,
                     const unsigned short* __restrict__ Wqt,
                     unsigned short* __restrict__ Qb,
                     unsigned short* __restrict__ Kb,
                     unsigned short* __restrict__ Vtb) {
  const int rowBase = blockIdx.y * 128, colBase = blockIdx.x * 64;
  v8f acc[2][2] = {};
  gemm_core<DMODEL>(Xb, Wqt, rowBase, colBase, 3 * DMODEL, acc);

  const int wave = threadIdx.x >> 5, lane = threadIdx.x & 31, ln = lane & 15;
  const int hi = lane >= 16, wm = wave >> 1, wn = wave & 1;
#pragma unroll
  for (int mi = 0; mi < 2; ++mi)
#pragma unroll
    for (int ni = 0; ni < 2; ++ni)
#pragma unroll
      for (int r = 0; r < 8; ++r) {
        int row = rowBase + wm * 32 + mi * 16 + r + (hi ? 8 : 0); // = t*B + b
        int col = colBase + wn * 32 + ni * 16 + ln;               // [0,3072)
        unsigned short us = f2bf(acc[mi][ni][r]);
        int t = row >> 3, bb = row & 7;
        int which = col >> 10, cc = col & 1023, n = cc >> 6, d = cc & 63;
        size_t hb = (size_t)(bb * NH + n);
        if (which == 0)      Qb[(hb * T_SEQ + t) * DH + d] = us;
        else if (which == 1) Kb[(hb * T_SEQ + t) * DH + d] = us;
        else                 Vtb[(hb * DH + d) * T_SEQ + t] = us;
      }
}

// --------------------------------------------------------------- output GEMM
__global__ __launch_bounds__(256)
void gemm_out_kernel(const unsigned short* __restrict__ Ab,
                     const unsigned short* __restrict__ Wot,
                     float* __restrict__ out) {
  const int rowBase = blockIdx.y * 128, colBase = blockIdx.x * 64;
  v8f acc[2][2] = {};
  gemm_core<DMODEL>(Ab, Wot, rowBase, colBase, DMODEL, acc);

  const int wave = threadIdx.x >> 5, lane = threadIdx.x & 31, ln = lane & 15;
  const int hi = lane >= 16, wm = wave >> 1, wn = wave & 1;
#pragma unroll
  for (int mi = 0; mi < 2; ++mi)
#pragma unroll
    for (int ni = 0; ni < 2; ++ni)
#pragma unroll
      for (int r = 0; r < 8; ++r) {
        int row = rowBase + wm * 32 + mi * 16 + r + (hi ? 8 : 0);
        int col = colBase + wn * 32 + ni * 16 + ln;
        out[(size_t)row * DMODEL + col] = acc[mi][ni][r];
      }
}

// ------------------------------------------------- two-pass flash attention
// Block: 128 threads (4 waves), each wave owns 16 query rows of a 64-row block.
template <bool PASS2>
__global__ __launch_bounds__(128)
void attn_kernel(const unsigned short* __restrict__ Qb,
                 const unsigned short* __restrict__ Kb,
                 const unsigned short* __restrict__ Vtb,
                 float* __restrict__ mbuf, float* __restrict__ lbuf,
                 unsigned short* __restrict__ Ab, float* __restrict__ cov) {
  __shared__ unsigned short Pl[4][16][64];
  const int iblk = blockIdx.x, n = blockIdx.y, b = blockIdx.z;
  const int wave = threadIdx.x >> 5, lane = threadIdx.x & 31, ln = lane & 15;
  const int hi = lane >= 16;
  const size_t head = (size_t)(b * NH + n);
  const unsigned short* Q  = Qb  + head * T_SEQ * DH;
  const unsigned short* K  = Kb  + head * T_SEQ * DH;
  const unsigned short* Vt = Vtb + head * DH * T_SEQ;
  const int ibase = iblk * 64, rloc = wave * 16;

  Frag16 qa[2];
#pragma unroll
  for (int kk = 0; kk < 2; ++kk) {
    int t = ibase + rloc + ln;
#pragma unroll
    for (int p = 0; p < 8; ++p)
      qa[kk].u[p] = *(const unsigned*)&Q[t * DH + kk * 32 + akp(p, hi)];
  }

  float m[8], l[8], linv[8];
#pragma unroll
  for (int r = 0; r < 8; ++r) { m[r] = -3.0e38f; l[r] = 0.0f; linv[r] = 0.0f; }
  if (PASS2) {
#pragma unroll
    for (int r = 0; r < 8; ++r) {
      int row = ibase + rloc + r + (hi ? 8 : 0);
      m[r]    = mbuf[head * T_SEQ + row];
      linv[r] = 1.0f / lbuf[head * T_SEQ + row];
    }
  }
  v8f oacc[4] = {};

  for (int jt = 0; jt <= iblk; ++jt) {
    v8f s[4] = {};
#pragma unroll
    for (int ni = 0; ni < 4; ++ni) {
      int j = jt * 64 + ni * 16 + ln;
#pragma unroll
      for (int kk = 0; kk < 2; ++kk) {
        Frag16 kf;
#pragma unroll
        for (int p = 0; p < 8; ++p)
          kf.u[p] = *(const unsigned*)&K[j * DH + kk * 32 + bkp(p, hi)];
        s[ni] = wmma_bf(qa[kk], kf, s[ni]);
      }
    }
    const bool diag = (jt == iblk);
#pragma unroll
    for (int ni = 0; ni < 4; ++ni)
#pragma unroll
      for (int r = 0; r < 8; ++r) {
        float sv = s[ni][r] * 0.125f;          // 1/sqrt(64)
        if (diag) {
          int i = ibase + rloc + r + (hi ? 8 : 0);
          int j = jt * 64 + ni * 16 + ln;
          if (j > i) sv = -3.0e38f;            // causal mask
        }
        s[ni][r] = sv;
      }

    if (!PASS2) {
#pragma unroll
      for (int r = 0; r < 8; ++r) {
        float tm = -3.0e38f;
#pragma unroll
        for (int ni = 0; ni < 4; ++ni) tm = fmaxf(tm, s[ni][r]);
#pragma unroll
        for (int off = 1; off < 16; off <<= 1)
          tm = fmaxf(tm, __shfl_xor(tm, off, 32));
        float nm = fmaxf(m[r], tm);
        float sum = 0.0f;
#pragma unroll
        for (int ni = 0; ni < 4; ++ni) sum += __expf(s[ni][r] - nm);
#pragma unroll
        for (int off = 1; off < 16; off <<= 1)
          sum += __shfl_xor(sum, off, 32);
        l[r] = l[r] * __expf(m[r] - nm) + sum;
        m[r] = nm;
      }
    } else {
#pragma unroll
      for (int ni = 0; ni < 4; ++ni)
#pragma unroll
        for (int r = 0; r < 8; ++r) {
          int i = ibase + rloc + r + (hi ? 8 : 0);
          int j = jt * 64 + ni * 16 + ln;
          float pv = __expf(s[ni][r] - m[r]) * linv[r];
          atomicAdd(&cov[((size_t)b * T_SEQ + j) * T_SEQ + i], pv * (1.0f / NH));
          Pl[wave][r + (hi ? 8 : 0)][ni * 16 + ln] = f2bf(pv);
        }
      __syncthreads();
#pragma unroll
      for (int kk = 0; kk < 2; ++kk) {
        Frag16 pf;
#pragma unroll
        for (int p = 0; p < 8; ++p)
          pf.u[p] = *(const unsigned*)&Pl[wave][ln][kk * 32 + akp(p, hi)];
#pragma unroll
        for (int dt = 0; dt < 4; ++dt) {
          Frag16 vf;
          int d = dt * 16 + ln;
#pragma unroll
          for (int p = 0; p < 8; ++p)
            vf.u[p] = *(const unsigned*)&Vt[d * T_SEQ + jt * 64 + kk * 32 + bkp(p, hi)];
          oacc[dt] = wmma_bf(pf, vf, oacc[dt]);
        }
      }
      __syncthreads();
    }
  }

  if (!PASS2) {
    if (ln == 0) {
#pragma unroll
      for (int r = 0; r < 8; ++r) {
        int row = ibase + rloc + r + (hi ? 8 : 0);
        mbuf[head * T_SEQ + row] = m[r];
        lbuf[head * T_SEQ + row] = l[r];
      }
    }
  } else {
#pragma unroll
    for (int dt = 0; dt < 4; ++dt)
#pragma unroll
      for (int r = 0; r < 8; ++r) {
        int i = ibase + rloc + r + (hi ? 8 : 0);
        int d = dt * 16 + ln;
        Ab[((size_t)i * BATCH + b) * DMODEL + n * DH + d] = f2bf(oacc[dt][r]);
      }
  }
}

// ------------------------------------------------------------------- launch
extern "C" void kernel_launch(void* const* d_in, const int* in_sizes, int n_in,
                              void* d_out, int out_size, void* d_ws, size_t ws_size,
                              hipStream_t stream) {
  (void)in_sizes; (void)n_in; (void)out_size; (void)ws_size;
  const float* input = (const float*)d_in[0];
  const float* pe    = (const float*)d_in[1];
  // d_in[2] = attn_mask: causal, applied analytically — not read
  const float* Wqkv  = (const float*)d_in[3];
  const float* Wo    = (const float*)d_in[4];

  char* ws = (char*)d_ws;
  size_t off = 0;
  auto alloc = [&](size_t bytes) -> void* {
    void* p = ws + off;
    off += (bytes + 255) & ~(size_t)255;
    return p;
  };
  unsigned short* Xb   = (unsigned short*)alloc((size_t)TB * DMODEL * 2);
  unsigned short* Wqt  = (unsigned short*)alloc((size_t)3 * DMODEL * DMODEL * 2);
  unsigned short* Wot  = (unsigned short*)alloc((size_t)DMODEL * DMODEL * 2);
  unsigned short* Qb   = (unsigned short*)alloc((size_t)BATCH * NH * T_SEQ * DH * 2);
  unsigned short* Kb   = (unsigned short*)alloc((size_t)BATCH * NH * T_SEQ * DH * 2);
  unsigned short* Vtb  = (unsigned short*)alloc((size_t)BATCH * NH * T_SEQ * DH * 2);
  unsigned short* Ab   = (unsigned short*)alloc((size_t)TB * DMODEL * 2);
  float*          mbuf = (float*)alloc((size_t)BATCH * NH * T_SEQ * 4);
  float*          lbuf = (float*)alloc((size_t)BATCH * NH * T_SEQ * 4);

  float* outAttn = (float*)d_out;                       // [T,B,1024]
  float* cov     = outAttn + (size_t)TB * DMODEL;       // [B,T,T]

  {
    int n = TB * DMODEL;
    pack_x_kernel<<<(n + 255) / 256, 256, 0, stream>>>(input, pe, Xb, n);
  }
  {
    int n = DMODEL * 3 * DMODEL;
    pack_wt_kernel<<<(n + 255) / 256, 256, 0, stream>>>(Wqkv, Wqt, DMODEL, 3 * DMODEL);
  }
  {
    int n = DMODEL * DMODEL;
    pack_wt_kernel<<<(n + 255) / 256, 256, 0, stream>>>(Wo, Wot, DMODEL, DMODEL);
  }
  {
    int n = BATCH * T_SEQ * T_SEQ;
    zero_kernel<<<(n + 255) / 256, 256, 0, stream>>>(cov, n);
  }
  gemm_qkv_kernel<<<dim3(3 * DMODEL / 64, TB / 128), 256, 0, stream>>>(
      Xb, Wqt, Qb, Kb, Vtb);
  attn_kernel<false><<<dim3(T_SEQ / 64, NH, BATCH), 128, 0, stream>>>(
      Qb, Kb, Vtb, mbuf, lbuf, Ab, cov);
  attn_kernel<true><<<dim3(T_SEQ / 64, NH, BATCH), 128, 0, stream>>>(
      Qb, Kb, Vtb, mbuf, lbuf, Ab, cov);
  gemm_out_kernel<<<dim3(DMODEL / 64, TB / 128), 256, 0, stream>>>(Ab, Wot, outAttn);
}
